// Tokenizer_51127290692385
// MI455X (gfx1250) — matrix operand order repaired and checked by reading, not compiled, not run on hardware
//
#include <hip/hip_runtime.h>
#include <hip/hip_bf16.h>
#include <cstddef>
#include <cstdint>

// Problem constants (from the reference)
#define BDIM 8
#define TDIM 2048
#define CDIM 256
#define VDIM 1024
#define BT   (BDIM * TDIM)          // 16384
#define NROWTILES (BT / 16)         // 1024 workgroups in the main kernel
#define SROW 1028                   // padded LDS row stride (floats) -> conflict-free

typedef __bf16 v16bf __attribute__((ext_vector_type(16)));
typedef __bf16 v8bf  __attribute__((ext_vector_type(8)));
typedef float  v8f   __attribute__((ext_vector_type(8)));

// ---------------------------------------------------------------------------
// Prep 1: split z (f32) into bf16 hi/lo pair (hi + lo ~= f32 value)
// ---------------------------------------------------------------------------
__global__ __launch_bounds__(256) void split_z_kernel(const float* __restrict__ z,
                                                      __bf16* __restrict__ zhi,
                                                      __bf16* __restrict__ zlo,
                                                      int n) {
    int i = blockIdx.x * 256 + threadIdx.x;
    if (i < n) {
        float v = z[i];
        __bf16 h = (__bf16)v;
        zhi[i] = h;
        zlo[i] = (__bf16)(v - (float)h);
    }
}

// ---------------------------------------------------------------------------
// Prep 2: split codebook (V,C) into bf16 hi/lo, kept in (V,C) layout, and ||e_v||^2.
// grid = V blocks, block = C = 256 threads.
// ---------------------------------------------------------------------------
__global__ __launch_bounds__(256) void prep_codebook_kernel(const float* __restrict__ e,
                                                            __bf16* __restrict__ eThi,
                                                            __bf16* __restrict__ eTlo,
                                                            float* __restrict__ en) {
    __shared__ float red[256];
    const int v = blockIdx.x;
    const int c = threadIdx.x;
    float val = e[(size_t)v * CDIM + c];
    __bf16 h = (__bf16)val;
    eThi[(size_t)v * CDIM + c] = h;
    eTlo[(size_t)v * CDIM + c] = (__bf16)(val - (float)h);
    red[c] = val * val;
    __syncthreads();
    for (int s = 128; s > 0; s >>= 1) {
        if (c < s) red[c] += red[c + s];
        __syncthreads();
    }
    if (c == 0) en[v] = red[0];
}

// ---------------------------------------------------------------------------
// Main fused kernel: s = ||e||^2 - 2 z@e^T via bf16x3 WMMA, then per-row
// argmin + softmax normalizer + histogram partial + e[idx] gather + commit.
// grid = 1024 (16 rows each), block = 256 threads = 8 waves (wave32).
// A fragments (z rows) are hoisted into registers once per wave; the inner
// loop issues only B loads + WMMA. Dynamic LDS: s_tile[16 * SROW] floats.
// ---------------------------------------------------------------------------
__global__ __launch_bounds__(256) void vq_main_kernel(
    const float*  __restrict__ z,
    const float*  __restrict__ codebook,
    const float*  __restrict__ mask,
    const __bf16* __restrict__ zhi,
    const __bf16* __restrict__ zlo,
    const __bf16* __restrict__ eThi,
    const __bf16* __restrict__ eTlo,
    const float*  __restrict__ en,
    float* __restrict__ out_zq,       // (BT, C)
    float* __restrict__ out_idx_f,    // (BT,)
    int*   __restrict__ idx_i32,      // workspace copy of indices
    float* __restrict__ hist_part,    // (NROWTILES, V) partial histograms
    float* __restrict__ commit_part)  // (NROWTILES,) partial commitment sums
{
    extern __shared__ float s_tile[];             // 16 * SROW floats (~64.25 KB)
    __shared__ float sh_m[16], sh_invZ[16], sh_mask[16];
    __shared__ int   sh_idx[16];
    __shared__ float sh_c[8];

    const int tid  = threadIdx.x;
    const int lane = tid & 31;
    const int wave = tid >> 5;                    // 0..7
    const int row0 = blockIdx.x * 16;

    const int nmod  = lane & 15;                  // N (and A's M) index within tile
    const int half  = lane >> 4;

    // ---------------- hoist all A fragments (K = 0..255) into registers -----
    // A fragment layout (ISA 7.12.2, 16-bit A 16x32): M = lane&15;
    // elements 0..7  -> K = c0 + 8*half + 0..7
    // elements 8..15 -> K = c0 + 8*half + 16..23
    v16bf a_hi[8], a_lo[8];
#pragma unroll
    for (int c = 0; c < 8; ++c) {
        const size_t abase = (size_t)(row0 + nmod) * CDIM + c * 32 + half * 8;
        v8bf ah0 = *reinterpret_cast<const v8bf*>(zhi + abase);
        v8bf ah1 = *reinterpret_cast<const v8bf*>(zhi + abase + 16);
        v8bf al0 = *reinterpret_cast<const v8bf*>(zlo + abase);
        v8bf al1 = *reinterpret_cast<const v8bf*>(zlo + abase + 16);
        a_hi[c] = __builtin_shufflevector(ah0, ah1, 0,1,2,3,4,5,6,7,8,9,10,11,12,13,14,15);
        a_lo[c] = __builtin_shufflevector(al0, al1, 0,1,2,3,4,5,6,7,8,9,10,11,12,13,14,15);
    }

    // ---------------- GEMM phase: 8 column tiles per wave -------------------
    for (int i = 0; i < 8; ++i) {
        const int v0 = (wave * 8 + i) * 16;       // codebook tile base (0..1008)

        // prefetch next tile's B stream into cache while this tile computes
        if (i < 7) {
            const size_t pf = (size_t)lane * VDIM + (v0 + 16);
            __builtin_prefetch(eThi + pf, 0, 1);
            __builtin_prefetch(eTlo + pf, 0, 1);
        }

        v8f acc = {};                             // f32 16x16 accumulator
#pragma unroll
        for (int c = 0; c < 8; ++c) {
            // B fragment: N = lane&15, K order same as A -> e[(v0+N)*C + c0 + 8*half + {0..7,16..23}]
            const size_t bbase = (size_t)(v0 + nmod) * CDIM + c * 32 + half * 8;
            v8bf bh0 = *reinterpret_cast<const v8bf*>(eThi + bbase);
            v8bf bh1 = *reinterpret_cast<const v8bf*>(eThi + bbase + 16);
            v8bf bl0 = *reinterpret_cast<const v8bf*>(eTlo + bbase);
            v8bf bl1 = *reinterpret_cast<const v8bf*>(eTlo + bbase + 16);
            v16bf b_hi = __builtin_shufflevector(bh0, bh1, 0,1,2,3,4,5,6,7,8,9,10,11,12,13,14,15);
            v16bf b_lo = __builtin_shufflevector(bl0, bl1, 0,1,2,3,4,5,6,7,8,9,10,11,12,13,14,15);

            // bf16x3: hi*hi + hi*lo + lo*hi, accumulated in f32
            acc = __builtin_amdgcn_wmma_f32_16x16x32_bf16(false, a_hi[c], false, b_hi,
                                                          (short)0, acc, false, false);
            acc = __builtin_amdgcn_wmma_f32_16x16x32_bf16(false, a_hi[c], false, b_lo,
                                                          (short)0, acc, false, false);
            acc = __builtin_amdgcn_wmma_f32_16x16x32_bf16(false, a_lo[c], false, b_hi,
                                                          (short)0, acc, false, false);
        }

        // D layout: VGPR r holds D[M = r + 8*half][N = lane&15]
        const int v = v0 + nmod;
        const float env = en[v];
#pragma unroll
        for (int r = 0; r < 8; ++r) {
            const int m = r + 8 * half;
            s_tile[m * SROW + v] = env - 2.0f * acc[r];
        }
    }
    __syncthreads();

    // ---------------- per-row argmin + softmax normalizer -------------------
    for (int rr = 0; rr < 2; ++rr) {
        const int r = wave * 2 + rr;              // 8 waves x 2 rows = 16 rows
        const float* srow = s_tile + r * SROW;

        float best = 3.4e38f;
        int   bidx = 0;
        for (int v = lane; v < VDIM; v += 32) {
            float sv = srow[v];
            if (sv < best) { best = sv; bidx = v; }
        }
        for (int off = 16; off > 0; off >>= 1) {
            float ov = __shfl_xor(best, off, 32);
            int   oi = __shfl_xor(bidx, off, 32);
            if (ov < best || (ov == best && oi < bidx)) { best = ov; bidx = oi; }
        }
        // all lanes now hold (min, first-argmin). softmax normalizer of -s:
        float zsum = 0.0f;
        for (int v = lane; v < VDIM; v += 32) zsum += __expf(best - srow[v]);
        for (int off = 16; off > 0; off >>= 1) zsum += __shfl_xor(zsum, off, 32);

        if (lane == 0) {
            sh_m[r]    = best;
            sh_idx[r]  = bidx;
            sh_invZ[r] = 1.0f / zsum;
            sh_mask[r] = mask[row0 + r];
            out_idx_f[row0 + r] = (float)bidx;
            idx_i32[row0 + r]   = bidx;
        }
    }
    __syncthreads();

    // ---------------- histogram partials (deterministic, no atomics) --------
#pragma unroll
    for (int q = 0; q < 4; ++q) {
        const int v = tid + q * 256;
        float h = 0.0f;
#pragma unroll
        for (int r = 0; r < 16; ++r)
            h += sh_mask[r] * sh_invZ[r] * __expf(sh_m[r] - s_tile[r * SROW + v]);
        hist_part[(size_t)blockIdx.x * VDIM + v] = h;
    }

    // ---------------- z_q gather + commitment partial ------------------------
    float csum = 0.0f;
    for (int rr = 0; rr < 2; ++rr) {
        const int r   = wave * 2 + rr;
        const int row = row0 + r;
        const float mk = sh_mask[r];
        const float* esrc = codebook + (size_t)sh_idx[r] * CDIM;
        for (int c = lane; c < CDIM; c += 32) {
            float zq = (mk == 0.0f) ? codebook[c] : esrc[c] * mk;   // pad_e = e[0]
            out_zq[(size_t)row * CDIM + c] = zq;
            float diff = z[(size_t)row * CDIM + c] - zq;
            csum += diff * diff * mk;
        }
    }
    for (int off = 16; off > 0; off >>= 1) csum += __shfl_xor(csum, off, 32);
    if (lane == 0) sh_c[wave] = csum;
    __syncthreads();
    if (tid == 0) {
        float t = 0.0f;
#pragma unroll
        for (int w = 0; w < 8; ++w) t += sh_c[w];
        commit_part[blockIdx.x] = t;
    }
}

// ---------------------------------------------------------------------------
// Reduce histogram partials (fixed order -> deterministic). grid = V.
// ---------------------------------------------------------------------------
__global__ __launch_bounds__(256) void hist_reduce_kernel(const float* __restrict__ hist_part,
                                                          float* __restrict__ hist) {
    __shared__ float red[256];
    const int v = blockIdx.x;
    float s = 0.0f;
    for (int blk = threadIdx.x; blk < NROWTILES; blk += 256)
        s += hist_part[(size_t)blk * VDIM + v];
    red[threadIdx.x] = s;
    __syncthreads();
    for (int st = 128; st > 0; st >>= 1) {
        if (threadIdx.x < st) red[threadIdx.x] += red[threadIdx.x + st];
        __syncthreads();
    }
    if (threadIdx.x == 0) hist[v] = red[0];
}

// ---------------------------------------------------------------------------
// Dedup scan: per batch row, keep-flags + block prefix sum -> source map.
// grid = B, block = 256 (each thread owns 8 consecutive positions).
// ---------------------------------------------------------------------------
__global__ __launch_bounds__(256) void dedup_scan_kernel(const int* __restrict__ idx_i32,
                                                         int* __restrict__ srcmap,
                                                         int* __restrict__ kcount) {
    __shared__ int tsum[256];
    const int b   = blockIdx.x;
    const int* row = idx_i32 + (size_t)b * TDIM;
    const int t0  = threadIdx.x * 8;

    int keep[8];
    int cnt = 0;
#pragma unroll
    for (int k = 0; k < 8; ++k) {
        const int t = t0 + k;
        const int kp = (t == 0) ? 1 : (row[t] != row[t - 1]);
        keep[k] = kp;
        cnt += kp;
    }
    tsum[threadIdx.x] = cnt;
    __syncthreads();
    // Hillis-Steele inclusive scan over thread totals
    for (int off = 1; off < 256; off <<= 1) {
        int add = (threadIdx.x >= off) ? tsum[threadIdx.x - off] : 0;
        __syncthreads();
        tsum[threadIdx.x] += add;
        __syncthreads();
    }
    int rank = tsum[threadIdx.x] - cnt;           // exclusive prefix
#pragma unroll
    for (int k = 0; k < 8; ++k) {
        if (keep[k]) { srcmap[(size_t)b * TDIM + rank] = t0 + k; ++rank; }
    }
    if (threadIdx.x == 255) kcount[b] = tsum[255];
}

// ---------------------------------------------------------------------------
// Dedup gather: z_q_disc[b, j] = j < K[b] ? z_q[b, src[j]] : e[0].
// grid = B*T blocks, block = C threads.
// ---------------------------------------------------------------------------
__global__ __launch_bounds__(256) void dedup_gather_kernel(const float* __restrict__ zq,
                                                           const float* __restrict__ codebook,
                                                           const int* __restrict__ srcmap,
                                                           const int* __restrict__ kcount,
                                                           float* __restrict__ zqdisc) {
    const int blk = blockIdx.x;
    const int b = blk >> 11;                      // /2048
    const int j = blk & (TDIM - 1);
    const int K = kcount[b];
    float val;
    if (j < K) {
        const int t = srcmap[(size_t)b * TDIM + j];
        val = zq[((size_t)b * TDIM + t) * CDIM + threadIdx.x];
    } else {
        val = codebook[threadIdx.x];              // pad_e = e[0]
    }
    zqdisc[(size_t)blk * CDIM + threadIdx.x] = val;
}

// ---------------------------------------------------------------------------
// Finalize: commitment loss and diversity loss (fixed-order reductions).
// grid = 1, block = 256.
// ---------------------------------------------------------------------------
__global__ __launch_bounds__(256) void finalize_kernel(const float* __restrict__ hist,
                                                       const float* __restrict__ commit_part,
                                                       const float* __restrict__ mask,
                                                       float* __restrict__ out) {
    __shared__ float red[256];
    const int tid = threadIdx.x;

    // sum(mask)
    float msum = 0.0f;
    for (int i = tid; i < BT; i += 256) msum += mask[i];
    red[tid] = msum; __syncthreads();
    for (int s = 128; s > 0; s >>= 1) { if (tid < s) red[tid] += red[tid + s]; __syncthreads(); }
    const float masksum = red[0];
    __syncthreads();

    // sum of commitment partials
    float cs = 0.0f;
    for (int i = tid; i < NROWTILES; i += 256) cs += commit_part[i];
    red[tid] = cs; __syncthreads();
    for (int s = 128; s > 0; s >>= 1) { if (tid < s) red[tid] += red[tid + s]; __syncthreads(); }
    const float commit = red[0];
    __syncthreads();

    // hist total
    float hs = 0.0f;
    for (int i = tid; i < VDIM; i += 256) hs += hist[i];
    red[tid] = hs; __syncthreads();
    for (int s = 128; s > 0; s >>= 1) { if (tid < s) red[tid] += red[tid + s]; __syncthreads(); }
    const float total = red[0];
    __syncthreads();

    // diversity = sum p * log(p*V + 1e-8)
    const float inv = 1.0f / (total + 1e-8f);
    float ds = 0.0f;
    for (int i = tid; i < VDIM; i += 256) {
        float p = hist[i] * inv;
        ds += p * __logf(p * (float)VDIM + 1e-8f);
    }
    red[tid] = ds; __syncthreads();
    for (int s = 128; s > 0; s >>= 1) { if (tid < s) red[tid] += red[tid + s]; __syncthreads(); }

    if (tid == 0) {
        out[0] = commit / (masksum * (float)CDIM);
        out[1] = red[0];
    }
}

// ---------------------------------------------------------------------------
// Host-side launcher
// ---------------------------------------------------------------------------
extern "C" void kernel_launch(void* const* d_in, const int* in_sizes, int n_in,
                              void* d_out, int out_size, void* d_ws, size_t ws_size,
                              hipStream_t stream) {
    (void)in_sizes; (void)n_in; (void)out_size; (void)ws_size;

    const float* z        = (const float*)d_in[0];   // (B,T,C)
    const float* codebook = (const float*)d_in[1];   // (V,C)
    const float* mask     = (const float*)d_in[2];   // (B,T,1)

    // d_out layout (flat tuple, return order):
    float* out      = (float*)d_out;
    float* out_zq   = out + 2;                                   // (B,T,C)
    float* out_zqd  = out + 2 + (size_t)BT * CDIM;               // (B,T,C)
    float* out_idx  = out + 2 + 2 * (size_t)BT * CDIM;           // (B,T)

    // workspace carve-up
    char* ws = (char*)d_ws;
    __bf16* zhi  = (__bf16*)ws;                                  // BT*C bf16
    __bf16* zlo  = zhi + (size_t)BT * CDIM;
    __bf16* eThi = zlo + (size_t)BT * CDIM;                      // C*V bf16, (V,C) row-major as e
    __bf16* eTlo = eThi + (size_t)CDIM * VDIM;
    float* en          = (float*)(eTlo + (size_t)CDIM * VDIM);   // V floats
    float* hist        = en + VDIM;                              // V floats
    float* hist_part   = hist + VDIM;                            // NROWTILES*V floats (4 MB)
    float* commit_part = hist_part + (size_t)NROWTILES * VDIM;   // NROWTILES floats
    int*   idx_i32     = (int*)(commit_part + NROWTILES);        // BT ints
    int*   srcmap      = idx_i32 + BT;                           // BT ints
    int*   kcount      = srcmap + BT;                            // B ints

    // 1) split z into bf16 hi/lo
    split_z_kernel<<<(BT * CDIM + 255) / 256, 256, 0, stream>>>(z, zhi, zlo, BT * CDIM);

    // 2) split codebook (kept (V,C)), compute ||e||^2
    prep_codebook_kernel<<<VDIM, 256, 0, stream>>>(codebook, eThi, eTlo, en);

    // 3) main fused WMMA kernel (64.25 KB dynamic LDS for the s tile)
    const size_t smem = (size_t)16 * SROW * sizeof(float);
    (void)hipFuncSetAttribute(reinterpret_cast<const void*>(vq_main_kernel),
                              hipFuncAttributeMaxDynamicSharedMemorySize, (int)smem);
    vq_main_kernel<<<NROWTILES, 256, smem, stream>>>(z, codebook, mask,
                                                     zhi, zlo, eThi, eTlo, en,
                                                     out_zq, out_idx, idx_i32,
                                                     hist_part, commit_part);

    // 4) deterministic histogram reduction
    hist_reduce_kernel<<<VDIM, 256, 0, stream>>>(hist_part, hist);

    // 5) run-length dedup: scan then gather
    dedup_scan_kernel<<<BDIM, 256, 0, stream>>>(idx_i32, srcmap, kcount);
    dedup_gather_kernel<<<BT, 256, 0, stream>>>(out_zq, codebook, srcmap, kcount, out_zqd);

    // 6) scalar losses
    finalize_kernel<<<1, 256, 0, stream>>>(hist, commit_part, mask, out);
}
